// RNN_LSTM_22703197127169
// MI455X (gfx1250) — matrix operand order repaired and checked by reading, not compiled, not run on hardware
//
#include <hip/hip_runtime.h>
#include <hip/hip_bf16.h>
#include <math.h>

#define Bz 32
#define Tz 512
#define Dz 256
#define Uz 512
#define G4 (4*Uz)      // 2048
#define NWG 8
#define KCHUNK 128
#define SMEME (256*KCHUNK)   // elements per LDS buffer (256 cols x 128 k)

typedef __bf16 bf16;
typedef __attribute__((ext_vector_type(16))) __bf16 v16bf;
typedef __attribute__((ext_vector_type(8)))  __bf16 v8bf;
typedef __attribute__((ext_vector_type(8)))  float  v8f;

union V16U { v16bf v; v8bf h[2]; };

// ---- WMMA fragment helpers (wave32, 16x16x32 bf16, fp32 acc) ----
// A (16x32, row-major, k contiguous): lane l holds row m=l&15,
// K range [g*16, g*16+16) with g=l>>4.
__device__ __forceinline__ v16bf load_A32(const bf16* base, int lda, int k0) {
  const int lane = threadIdx.x & 31;
  const int m = lane & 15, g = lane >> 4;
  const bf16* p = base + (size_t)m * lda + k0 + g * 16;
  V16U u;
  u.h[0] = *(const v8bf*)(p);
  u.h[1] = *(const v8bf*)(p + 8);
  return u.v;
}
// B (32x16) from N-major weights Bt[n][k] in global memory.
__device__ __forceinline__ v16bf load_B32(const bf16* bt, int ldb, int n0, int k0) {
  const int lane = threadIdx.x & 31;
  const int n = lane & 15, g = lane >> 4;
  const bf16* p = bt + (size_t)(n0 + n) * ldb + k0 + g * 16;
  V16U u;
  u.h[0] = *(const v8bf*)(p);
  u.h[1] = *(const v8bf*)(p + 8);
  return u.v;
}
// B (32x16) from LDS-staged chunk: sbuf[colLocal][0..KCHUNK)
__device__ __forceinline__ v16bf load_B_lds(const bf16* sbuf, int colbase, int kk) {
  const int lane = threadIdx.x & 31;
  const int n = lane & 15, g = lane >> 4;
  const bf16* p = sbuf + (size_t)(colbase + n) * KCHUNK + kk + g * 16;
  V16U u;
  u.h[0] = *(const v8bf*)(p);
  u.h[1] = *(const v8bf*)(p + 8);
  return u.v;
}
__device__ __forceinline__ v8f wmma_bf16(v16bf a, v16bf b, v8f c) {
  return __builtin_amdgcn_wmma_f32_16x16x32_bf16(false, a, false, b, (short)0, c, false, false);
}

__device__ __forceinline__ float sigf(float x) { return 1.0f / (1.0f + __expf(-x)); }

// ---- CDNA5 async global->LDS copy (ASYNCcnt-tracked) ----
__device__ __forceinline__ void async_b128(unsigned lds_addr, const bf16* gptr) {
  asm volatile("global_load_async_to_lds_b128 %0, %1, off" :: "v"(lds_addr), "v"(gptr) : "memory");
}
__device__ __forceinline__ void wait_async_le16() {
  asm volatile("s_wait_asynccnt 0x10" ::: "memory");
}
__device__ __forceinline__ void wait_async_le0() {
  asm volatile("s_wait_asynccnt 0x0" ::: "memory");
}

// Stage one chunk: 256 gate-columns x KCHUNK k of zt into sbuf.
// thread tid owns colLocal = tid (gate = tid>>6, rest = tid&63).
// Per wave: 16 async_b128 issues (32 lanes x 16B each = 8KB/wave, 64KB/WG).
__device__ __forceinline__ void issue_chunk(const bf16* zt, int ldb, int wg,
                                            bf16* sbuf, int ztk0) {
  const int tid = threadIdx.x;
  const int gate = tid >> 6, rest = tid & 63;
  const bf16* gp = zt + (size_t)(gate * Uz + wg * 64 + rest) * ldb + ztk0;
  unsigned lp = (unsigned)(size_t)sbuf + (unsigned)tid * (KCHUNK * 2);
#pragma unroll
  for (int part = 0; part < KCHUNK / 8; ++part)
    async_b128(lp + part * 16, gp + part * 8);
}

// ---- global (8-block) sense barrier: bar[0]=count, bar[1]=generation ----
__device__ __forceinline__ void gbar(unsigned* bar) {
  __syncthreads();
  if (threadIdx.x == 0) {
    __threadfence();
    unsigned gen = __hip_atomic_load(bar + 1, __ATOMIC_RELAXED, __HIP_MEMORY_SCOPE_AGENT);
    unsigned a = __hip_atomic_fetch_add(bar, 1u, __ATOMIC_ACQ_REL, __HIP_MEMORY_SCOPE_AGENT);
    if (a == NWG - 1) {
      __hip_atomic_store(bar, 0u, __ATOMIC_RELAXED, __HIP_MEMORY_SCOPE_AGENT);
      __hip_atomic_store(bar + 1, gen + 1u, __ATOMIC_RELEASE, __HIP_MEMORY_SCOPE_AGENT);
    } else {
      while (__hip_atomic_load(bar + 1, __ATOMIC_ACQUIRE, __HIP_MEMORY_SCOPE_AGENT) == gen) {
        __builtin_amdgcn_s_sleep(1);
      }
    }
  }
  __syncthreads();
  __threadfence();
}

// ---- prep kernels ----
__global__ void k_cvt_x(const float* __restrict__ x, bf16* __restrict__ xb, int n) {
  for (int i = blockIdx.x * blockDim.x + threadIdx.x; i < n; i += gridDim.x * blockDim.x)
    xb[i] = (bf16)x[i];
}

// Zt[n][k] = (k < inD ? W[k][n] : R[k-inD][n]) as bf16; W,R row-major [k][2048]
__global__ void k_build_zt(const float* __restrict__ W, const float* __restrict__ R,
                           bf16* __restrict__ zt, int inD, int inK) {
  const size_t total = (size_t)G4 * inK;
  for (size_t i = blockIdx.x * (size_t)blockDim.x + threadIdx.x; i < total;
       i += (size_t)gridDim.x * blockDim.x) {
    int n = (int)(i / inK);
    int k = (int)(i % inK);
    float v = (k < inD) ? W[(size_t)k * G4 + n] : R[(size_t)(k - inD) * G4 + n];
    zt[i] = (bf16)v;
  }
}

// Wdt[n][k] = Wd[k][n], n<256, k<512
__global__ void k_build_wdt(const float* __restrict__ Wd, bf16* __restrict__ wdt) {
  const int total = Dz * Uz;
  for (int i = blockIdx.x * blockDim.x + threadIdx.x; i < total; i += gridDim.x * blockDim.x) {
    int n = i / Uz, k = i % Uz;
    wdt[i] = (bf16)Wd[(size_t)k * Dz + n];
  }
}

__global__ void k_init_state(const float* h0, const float* c0, const float* h1, const float* c1,
                             const float* h2, const float* c2,
                             bf16* __restrict__ hstate, float* __restrict__ cbuf,
                             unsigned* __restrict__ bar) {
  const int HS = 3 * Bz * Uz;
  for (int i = blockIdx.x * blockDim.x + threadIdx.x; i < HS; i += gridDim.x * blockDim.x) {
    int l = i / (Bz * Uz);
    int u = i % Uz;
    const float* hp = (l == 0) ? h0 : (l == 1) ? h1 : h2;
    const float* cp = (l == 0) ? c0 : (l == 1) ? c1 : c2;
    bf16 hv = (bf16)hp[u];
    hstate[i] = hv;            // parity 0
    hstate[HS + i] = hv;       // parity 1 (overwritten before use)
    cbuf[i] = cp[u];
  }
  if (blockIdx.x == 0 && threadIdx.x == 0) { bar[0] = 0u; bar[1] = 0u; }
}

// ---- X0W precompute: x0wt[t][n][m] = sum_k x[m][t][k] * W0[k][n] + b0[n] ----
__global__ void k_xw(const bf16* __restrict__ xb, const bf16* __restrict__ zt0,
                     const float* __restrict__ b0, float* __restrict__ x0wt) {
  const int wave = threadIdx.x >> 5;
  const int id = blockIdx.x * 8 + wave;          // 512*128*2 = 131072 tiles
  const int mt = id & 1;
  const int nt = (id >> 1) & 127;
  const int t  = id >> 8;
  const int lane = threadIdx.x & 31;
  const int g = lane >> 4, nl = lane & 15;

  const bf16* abase = xb + (size_t)(mt * 16) * (Tz * Dz) + (size_t)t * Dz;
  v8f acc = {0.f, 0.f, 0.f, 0.f, 0.f, 0.f, 0.f, 0.f};
  for (int k0 = 0; k0 < Dz; k0 += 32) {
    v16bf a = load_A32(abase, Tz * Dz, k0);
    v16bf b = load_B32(zt0, Dz + Uz, nt * 16, k0);
    acc = wmma_bf16(a, b, acc);
  }
  const int n = nt * 16 + nl;
  const float bias = b0[n];
#pragma unroll
  for (int v = 0; v < 8; ++v) acc[v] += bias;
  *(v8f*)(x0wt + ((size_t)t * G4 + n) * Bz + mt * 16 + g * 8) = acc;
}

// ---- persistent sequential LSTM scan with async-LDS weight pipeline ----
template <bool PRE>
__global__ void __launch_bounds__(256, 1)
k_seq(const bf16* __restrict__ xb,
      const bf16* __restrict__ zt0, const bf16* __restrict__ zt1, const bf16* __restrict__ zt2,
      const bf16* __restrict__ wdt,
      const float* __restrict__ b0, const float* __restrict__ b1, const float* __restrict__ b2,
      const float* __restrict__ bd,
      const float* __restrict__ x0wt,
      bf16* __restrict__ hstate, float* __restrict__ cbuf,
      float* __restrict__ out, unsigned* __restrict__ bar) {
  extern __shared__ __align__(16) bf16 smem[];   // 2 * SMEME bf16 = 128KB
  const int wg = blockIdx.x;
  const int wave = threadIdx.x >> 5;
  const int lane = threadIdx.x & 31;
  const int mt = wave & 1;        // batch tile (rows 16*mt .. +16)
  const int nt = wave >> 1;       // unit sub-tile 0..3
  const int g = lane >> 4, nl = lane & 15;
  const int u0 = wg * 64 + nt * 16;   // this wave's 16-unit slice
  const int HS = 3 * Bz * Uz;

  for (int t = 0; t < Tz; ++t) {
    const int p = t & 1;
    bf16* hread = hstate + (size_t)p * HS;        // states from step t-1
    bf16* hwrit = hstate + (size_t)(p ^ 1) * HS;  // states produced at step t

#pragma unroll
    for (int l = 0; l < 3; ++l) {
      const bf16* zt = (l == 0) ? zt0 : (l == 1) ? zt1 : zt2;
      const int ldb = (l == 0) ? (Dz + Uz) : (2 * Uz);
      const int K0 = (l == 0) ? (PRE ? Dz : 0) : 0;     // first zt k staged
      const int nc = (l == 0) ? (PRE ? 4 : 6) : 8;      // chunks of KCHUNK
      const int S  = (l == 0) ? Dz : Uz;                // zt k < S -> input phase
      const bf16* ain = (l == 0)
          ? xb + (size_t)(mt * 16) * (Tz * Dz) + (size_t)t * Dz
          : hwrit + (size_t)(l - 1) * Bz * Uz + (size_t)(mt * 16) * Uz;
      const int ldain = (l == 0) ? (Tz * Dz) : Uz;
      const bf16* ar = hread + (size_t)l * Bz * Uz + (size_t)(mt * 16) * Uz;

      v8f a0, a1, a2, a3;
      if (l == 0 && PRE) {
        const float* xp = x0wt + ((size_t)t * G4 + u0 + nl) * Bz + mt * 16 + g * 8;
        a0 = *(const v8f*)(xp + (size_t)0 * Uz * Bz);
        a1 = *(const v8f*)(xp + (size_t)1 * Uz * Bz);
        a2 = *(const v8f*)(xp + (size_t)2 * Uz * Bz);
        a3 = *(const v8f*)(xp + (size_t)3 * Uz * Bz);
      } else {
        a0 = a1 = a2 = a3 = (v8f){0.f, 0.f, 0.f, 0.f, 0.f, 0.f, 0.f, 0.f};
      }

      // double-buffered async weight pipeline over k chunks
      issue_chunk(zt, ldb, wg, smem, K0);
      for (int ci = 0; ci < nc; ++ci) {
        if (ci + 1 < nc) {
          issue_chunk(zt, ldb, wg, smem + ((ci + 1) & 1) * SMEME, K0 + (ci + 1) * KCHUNK);
          wait_async_le16();           // chunk ci complete (in-order completion)
        } else {
          wait_async_le0();
        }
        __syncthreads();               // all waves' copies visible in LDS
        const bf16* sb = smem + (ci & 1) * SMEME;
        const int ztc = K0 + ci * KCHUNK;
        const bf16* ab  = (ztc < S) ? ain : ar;        // chunk never straddles phases
        const int  lda2 = (ztc < S) ? ldain : Uz;
        const int  ak0  = (ztc < S) ? ztc : (ztc - S);
#pragma unroll
        for (int kq = 0; kq < KCHUNK / 32; ++kq) {
          const int kk = kq * 32;
          v16bf a = load_A32(ab, lda2, ak0 + kk);
          a0 = wmma_bf16(a, load_B_lds(sb, 0 * 64 + nt * 16, kk), a0);
          a1 = wmma_bf16(a, load_B_lds(sb, 1 * 64 + nt * 16, kk), a1);
          a2 = wmma_bf16(a, load_B_lds(sb, 2 * 64 + nt * 16, kk), a2);
          a3 = wmma_bf16(a, load_B_lds(sb, 3 * 64 + nt * 16, kk), a3);
        }
        __syncthreads();               // done reading before buffer reuse
      }

      // bias (b0 folded into x0wt on the PRE path)
      if (l > 0 || !PRE) {
        const float* bl = (l == 0) ? b0 : (l == 1) ? b1 : b2;
        const float bi  = bl[0 * Uz + u0 + nl];
        const float bf_ = bl[1 * Uz + u0 + nl];
        const float bg  = bl[2 * Uz + u0 + nl];
        const float bo  = bl[3 * Uz + u0 + nl];
#pragma unroll
        for (int v = 0; v < 8; ++v) { a0[v] += bi; a1[v] += bf_; a2[v] += bg; a3[v] += bo; }
      }

      // elementwise cell update (this wave owns its (m,u) slice of c)
      float* cl = cbuf + (size_t)l * Bz * Uz;
      bf16* hl = hwrit + (size_t)l * Bz * Uz;
#pragma unroll
      for (int v = 0; v < 8; ++v) {
        const int m = mt * 16 + g * 8 + v;
        const int u = u0 + nl;
        const size_t idx = (size_t)m * Uz + u;
        const float cold = cl[idx];
        const float cn = sigf(a1[v]) * cold + sigf(a0[v]) * tanhf(a2[v]);
        const float hn = sigf(a3[v]) * tanhf(cn);
        cl[idx] = cn;
        hl[idx] = (bf16)hn;
      }
      gbar(bar);
    }

    // dense head: y = h2 @ Wd + bd  (32 tiles over 64 waves -> waves 0..3 of each WG)
    if (wave < 4) {
      v8f acc = {0.f, 0.f, 0.f, 0.f, 0.f, 0.f, 0.f, 0.f};
      const int dmt = wave & 1;
      const int ntile = wg * 2 + (wave >> 1);
      const bf16* ad = hwrit + (size_t)2 * Bz * Uz + (size_t)(dmt * 16) * Uz;
      for (int kk = 0; kk < Uz; kk += 32) {
        v16bf a = load_A32(ad, Uz, kk);
        acc = wmma_bf16(a, load_B32(wdt, Uz, ntile * 16, kk), acc);
      }
      const int n = ntile * 16 + nl;
      const float bias = bd[n];
#pragma unroll
      for (int v = 0; v < 8; ++v) {
        const int m = dmt * 16 + g * 8 + v;
        out[((size_t)m * Tz + t) * Dz + n] = acc[v] + bias;
      }
    }
  }
}

extern "C" void kernel_launch(void* const* d_in, const int* in_sizes, int n_in,
                              void* d_out, int out_size, void* d_ws, size_t ws_size,
                              hipStream_t stream) {
  (void)in_sizes; (void)n_in; (void)out_size;
  const float* x  = (const float*)d_in[0];
  const float* W0 = (const float*)d_in[1];
  const float* R0 = (const float*)d_in[2];
  const float* b0 = (const float*)d_in[3];
  const float* h0 = (const float*)d_in[4];
  const float* c0 = (const float*)d_in[5];
  const float* W1 = (const float*)d_in[6];
  const float* R1 = (const float*)d_in[7];
  const float* b1 = (const float*)d_in[8];
  const float* h1 = (const float*)d_in[9];
  const float* c1 = (const float*)d_in[10];
  const float* W2 = (const float*)d_in[11];
  const float* R2 = (const float*)d_in[12];
  const float* b2 = (const float*)d_in[13];
  const float* h2 = (const float*)d_in[14];
  const float* c2 = (const float*)d_in[15];
  const float* Wd = (const float*)d_in[16];
  const float* bd = (const float*)d_in[17];
  float* out = (float*)d_out;

  // carve workspace
  char* w = (char*)d_ws;
  size_t off = 0;
  auto take = [&](size_t bytes) -> void* {
    void* p = w + off;
    off += (bytes + 255) & ~(size_t)255;
    return p;
  };
  bf16*  xb     = (bf16*)  take((size_t)Bz * Tz * Dz * 2);
  bf16*  zt0    = (bf16*)  take((size_t)G4 * (Dz + Uz) * 2);
  bf16*  zt1    = (bf16*)  take((size_t)G4 * (2 * Uz) * 2);
  bf16*  zt2    = (bf16*)  take((size_t)G4 * (2 * Uz) * 2);
  bf16*  wdt    = (bf16*)  take((size_t)Dz * Uz * 2);
  bf16*  hstate = (bf16*)  take((size_t)2 * 3 * Bz * Uz * 2);
  float* cbuf   = (float*) take((size_t)3 * Bz * Uz * 4);
  unsigned* bar = (unsigned*) take(256);
  float* x0wt   = (float*) take((size_t)Tz * G4 * Bz * 4);
  const bool pre = (ws_size >= off);

  k_cvt_x<<<4096, 256, 0, stream>>>(x, xb, Bz * Tz * Dz);
  k_build_zt<<<2048, 256, 0, stream>>>(W0, R0, zt0, Dz, Dz + Uz);
  k_build_zt<<<4096, 256, 0, stream>>>(W1, R1, zt1, Uz, 2 * Uz);
  k_build_zt<<<4096, 256, 0, stream>>>(W2, R2, zt2, Uz, 2 * Uz);
  k_build_wdt<<<512, 256, 0, stream>>>(Wd, wdt);
  k_init_state<<<192, 256, 0, stream>>>(h0, c0, h1, c1, h2, c2, hstate, cbuf, bar);

  const size_t smem_bytes = (size_t)2 * SMEME * sizeof(bf16);  // 128KB
  if (pre) {
    k_xw<<<16384, 256, 0, stream>>>(xb, zt0, b0, x0wt);
    k_seq<true><<<NWG, 256, smem_bytes, stream>>>(xb, zt0, zt1, zt2, wdt, b0, b1, b2, bd,
                                                  x0wt, hstate, cbuf, out, bar);
  } else {
    k_seq<false><<<NWG, 256, smem_bytes, stream>>>(xb, zt0, zt1, zt2, wdt, b0, b1, b2, bd,
                                                   x0wt, hstate, cbuf, out, bar);
  }
}
